// GCMCLayer_42734924595923
// MI455X (gfx1250) — matrix operand order repaired and checked by reading, not compiled, not run on hardware
//
#include <hip/hip_runtime.h>

typedef __attribute__((ext_vector_type(8)))  float        v8f;
typedef __attribute__((ext_vector_type(16))) __bf16       v16bf;
typedef __attribute__((ext_vector_type(2)))  __bf16       bf16x2;
typedef __attribute__((ext_vector_type(4)))  __bf16       bf16x4;
typedef __attribute__((ext_vector_type(4)))  unsigned int u32x4;

union Frag16 {
    u32x4  q[2];
    v16bf  v;
};

#define R_NUM   5
#define NU_NUM  100000
#define NI_NUM  50000
#define DIN_NUM 512
#define K_NUM   128
#define MSG_NUM 640
#define OUT_NUM 256
#define E_NUM   200000

#define LDXU    100032   // NU rounded up to 64
#define LDXI    50048    // NI rounded up to 64

#define A_STRIDE 40   // bf16 elems per LDS A row (80B, keeps b128 alignment)
#define B_STRIDE 40   // bf16 elems per LDS B column (K-major)

// ---------------------------------------------------------------------------
// GEMM core: C[64 x 128] += A[64 x 32] * B[32 x 128] for one staged k-chunk.
// bf16 WMMA 16x16x32, f32 accumulation. Wave w owns cols [16w,16w+16).
// ---------------------------------------------------------------------------
__device__ __forceinline__ void wmma_tile_step(
    const __bf16 (*As)[A_STRIDE],   // [64][A_STRIDE]
    const __bf16 (*Bsf)[B_STRIDE],  // [128][B_STRIDE] K-major per column
    int wave, int l16, int hl, int kb,
    v8f acc[4])
{
    Frag16 bf;
    const int n = wave * 16 + l16;
    bf.q[0] = *(const u32x4*)&Bsf[n][hl * 8];
    bf.q[1] = *(const u32x4*)&Bsf[n][hl * 8 + 16];

    #pragma unroll
    for (int rt = 0; rt < 4; ++rt) {
        const int m = rt * 16 + l16;
        Frag16 af;
        af.q[0] = *(const u32x4*)&As[m][kb];        // elems 0..7  : K kb..kb+7
        af.q[1] = *(const u32x4*)&As[m][kb + 16];   // elems 8..15 : K kb+16..kb+23
        acc[rt] = __builtin_amdgcn_wmma_f32_16x16x32_bf16(
            false, af.v, false, bf.v, (short)0, acc[rt], false, false);
    }
}

// ---------------------------------------------------------------------------
// Kernel 1: x[r][n][:] = (feat[n] @ W[r]) * cj[n]   (bf16 out, padded stride)
// Double-buffered pipeline: one barrier / chunk, loads overlap WMMA.
// ---------------------------------------------------------------------------
__global__ __launch_bounds__(256)
void gcmc_xform(const float* __restrict__ feat,   // [nrows, 512]
                const float* __restrict__ W,      // [R, 512, 128]
                const float* __restrict__ cj,     // [nrows]
                __bf16* __restrict__ xout,        // [R, ldx, 128] bf16
                int nrows, int ldx)
{
    __shared__ __bf16 As[2][64][A_STRIDE];
    __shared__ __bf16 Bsf[2][128][B_STRIDE];

    const int r    = blockIdx.y;
    const int row0 = blockIdx.x * 64;
    const int tid  = threadIdx.x;
    const int lane = tid & 31;
    const int wave = tid >> 5;
    const int hl   = lane >> 4;
    const int l16  = lane & 15;
    const int kb   = hl * 8;

    v8f acc[4] = {};
    const float* Wr = W + (size_t)r * DIN_NUM * K_NUM;

    // Per-thread staging geometry (fixed across chunks)
    const int rowA = tid >> 3;             // A row within tile (+32 for i=1)
    const int c4a  = (tid & 7) * 4;        // A col group
    const int prow = 2 * (tid >> 5);       // B even row within chunk (+16 for i=1)
    const int c4b  = (tid & 31) * 4;       // B col group

    // Per-thread base pointers, advanced by constant stride per chunk
    const float* aptr[2];
    const float* bptr[2];
    #pragma unroll
    for (int i = 0; i < 2; ++i) {
        int grow = row0 + rowA + 32 * i;
        if (grow >= nrows) grow = nrows - 1;   // clamp, branch-free
        aptr[i] = &feat[(size_t)grow * DIN_NUM + c4a];
        bptr[i] = &Wr[(size_t)(prow + 16 * i) * K_NUM + c4b];
    }

    float4 fa[2], fb0[2], fb1[2];

    auto load_stage = [&]() {
        #pragma unroll
        for (int i = 0; i < 2; ++i) {
            fa[i] = *(const float4*)aptr[i];
            aptr[i] += 32;
        }
        #pragma unroll
        for (int i = 0; i < 2; ++i) {
            fb0[i] = *(const float4*)bptr[i];
            fb1[i] = *(const float4*)(bptr[i] + K_NUM);
            bptr[i] += 32 * K_NUM;
        }
    };
    auto store_stage = [&](int buf) {
        #pragma unroll
        for (int i = 0; i < 2; ++i) {
            bf16x4 b;
            b[0] = (__bf16)fa[i].x; b[1] = (__bf16)fa[i].y;
            b[2] = (__bf16)fa[i].z; b[3] = (__bf16)fa[i].w;
            *(bf16x4*)&As[buf][rowA + 32 * i][c4a] = b;
        }
        #pragma unroll
        for (int i = 0; i < 2; ++i) {          // K-pair packed b32 stores
            const int p2 = prow + 16 * i;
            bf16x2 t;
            t[0] = (__bf16)fb0[i].x; t[1] = (__bf16)fb1[i].x;
            *(bf16x2*)&Bsf[buf][c4b + 0][p2] = t;
            t[0] = (__bf16)fb0[i].y; t[1] = (__bf16)fb1[i].y;
            *(bf16x2*)&Bsf[buf][c4b + 1][p2] = t;
            t[0] = (__bf16)fb0[i].z; t[1] = (__bf16)fb1[i].z;
            *(bf16x2*)&Bsf[buf][c4b + 2][p2] = t;
            t[0] = (__bf16)fb0[i].w; t[1] = (__bf16)fb1[i].w;
            *(bf16x2*)&Bsf[buf][c4b + 3][p2] = t;
        }
    };

    load_stage();
    store_stage(0);
    __syncthreads();

    constexpr int NK = DIN_NUM / 32;   // 16
    #pragma unroll 2
    for (int kk = 0; kk < NK; ++kk) {
        const int buf = kk & 1;
        if (kk + 1 < NK) load_stage();                    // prefetch to regs
        wmma_tile_step(As[buf], Bsf[buf], wave, l16, hl, kb, acc);
        if (kk + 1 < NK) store_stage(buf ^ 1);            // fill other buffer
        __syncthreads();
    }

    // Epilogue: clamped cj loads, UNGUARDED stores (padded ldx rows).
    const int col = wave * 16 + l16;
    __bf16* xr = xout + (size_t)r * ldx * K_NUM;
    #pragma unroll
    for (int rt = 0; rt < 4; ++rt) {
        #pragma unroll
        for (int v = 0; v < 8; ++v) {
            int grow = row0 + rt * 16 + hl * 8 + v;
            int gc = (grow < nrows) ? grow : (nrows - 1);
            float s = cj[gc];
            xr[(size_t)grow * K_NUM + col] = (__bf16)(acc[rt][v] * s);
        }
    }
}

// ---------------------------------------------------------------------------
// Kernel 2: edge gather + scatter-add.  One wave per edge: lane j handles
// message slots [4j, 4j+4): 8-byte bf16 read, 4 f32 global atomic adds.
// ---------------------------------------------------------------------------
__global__ __launch_bounds__(256)
void gcmc_scatter(const __bf16* __restrict__ x,    // [R, ldsrc, 128] bf16
                  const int* __restrict__ esrc,    // [R, E]
                  const int* __restrict__ edst,    // [R, E]
                  float* __restrict__ h,           // [ndst, 640] f32
                  int ldsrc)
{
    const int gw = blockIdx.x * 8 + (threadIdx.x >> 5);
    if (gw >= R_NUM * E_NUM) return;   // whole-wave exit only
    const int lane = threadIdx.x & 31;
    const int r = gw / E_NUM;
    const int sidx = esrc[gw];
    const int didx = edst[gw];

    const bf16x4 m = *(const bf16x4*)(x + ((size_t)r * ldsrc + sidx) * K_NUM + lane * 4);
    float* dst = h + (size_t)didx * MSG_NUM + r * K_NUM + lane * 4;
    atomicAdd(dst + 0, (float)m[0]);
    atomicAdd(dst + 1, (float)m[1]);
    atomicAdd(dst + 2, (float)m[2]);
    atomicAdd(dst + 3, (float)m[3]);
}

// ---------------------------------------------------------------------------
// Kernel 3: out = ci[row] * (h @ Wfc) + b.  64 x 128 tile, 2 column halves.
// Same double-buffered pipeline; block-uniform tail guard.
// ---------------------------------------------------------------------------
__global__ __launch_bounds__(256)
void gcmc_fc(const float* __restrict__ h,     // [nrows, 640] f32
             const float* __restrict__ ci,    // [nrows]
             const float* __restrict__ Wfc,   // [640, 256]
             const float* __restrict__ bias,  // [256]
             float* __restrict__ out,         // [nrows, 256]
             int nrows)
{
    __shared__ __bf16 As[2][64][A_STRIDE];
    __shared__ __bf16 Bsf[2][128][B_STRIDE];

    const int row0    = blockIdx.x * 64;
    const int colbase = blockIdx.y * 128;
    const int tid  = threadIdx.x;
    const int lane = tid & 31;
    const int wave = tid >> 5;
    const int hl   = lane >> 4;
    const int l16  = lane & 15;
    const int kb   = hl * 8;

    v8f acc[4] = {};

    const int rowA = tid >> 3;
    const int c4a  = (tid & 7) * 4;
    const int prow = 2 * (tid >> 5);
    const int c4b  = (tid & 31) * 4;

    const float* aptr[2];
    const float* bptr[2];
    #pragma unroll
    for (int i = 0; i < 2; ++i) {
        int grow = row0 + rowA + 32 * i;
        if (grow >= nrows) grow = nrows - 1;
        aptr[i] = &h[(size_t)grow * MSG_NUM + c4a];
        bptr[i] = &Wfc[(size_t)(prow + 16 * i) * OUT_NUM + colbase + c4b];
    }

    float4 fa[2], fb0[2], fb1[2];

    auto load_stage = [&]() {
        #pragma unroll
        for (int i = 0; i < 2; ++i) {
            fa[i] = *(const float4*)aptr[i];
            aptr[i] += 32;
        }
        #pragma unroll
        for (int i = 0; i < 2; ++i) {
            fb0[i] = *(const float4*)bptr[i];
            fb1[i] = *(const float4*)(bptr[i] + OUT_NUM);
            bptr[i] += 32 * OUT_NUM;
        }
    };
    auto store_stage = [&](int buf) {
        #pragma unroll
        for (int i = 0; i < 2; ++i) {
            bf16x4 b;
            b[0] = (__bf16)fa[i].x; b[1] = (__bf16)fa[i].y;
            b[2] = (__bf16)fa[i].z; b[3] = (__bf16)fa[i].w;
            *(bf16x4*)&As[buf][rowA + 32 * i][c4a] = b;
        }
        #pragma unroll
        for (int i = 0; i < 2; ++i) {
            const int p2 = prow + 16 * i;
            bf16x2 t;
            t[0] = (__bf16)fb0[i].x; t[1] = (__bf16)fb1[i].x;
            *(bf16x2*)&Bsf[buf][c4b + 0][p2] = t;
            t[0] = (__bf16)fb0[i].y; t[1] = (__bf16)fb1[i].y;
            *(bf16x2*)&Bsf[buf][c4b + 1][p2] = t;
            t[0] = (__bf16)fb0[i].z; t[1] = (__bf16)fb1[i].z;
            *(bf16x2*)&Bsf[buf][c4b + 2][p2] = t;
            t[0] = (__bf16)fb0[i].w; t[1] = (__bf16)fb1[i].w;
            *(bf16x2*)&Bsf[buf][c4b + 3][p2] = t;
        }
    };

    load_stage();
    store_stage(0);
    __syncthreads();

    constexpr int NK = MSG_NUM / 32;   // 20
    #pragma unroll 2
    for (int kk = 0; kk < NK; ++kk) {
        const int buf = kk & 1;
        if (kk + 1 < NK) load_stage();
        wmma_tile_step(As[buf], Bsf[buf], wave, l16, hl, kb, acc);
        if (kk + 1 < NK) store_stage(buf ^ 1);
        __syncthreads();
    }

    // Epilogue: out = ci[row] * acc + bias[col]; block-uniform tail guard.
    const int col = colbase + wave * 16 + l16;
    const float b = bias[col];
    if (row0 + 64 <= nrows) {
        #pragma unroll
        for (int rt = 0; rt < 4; ++rt) {
            #pragma unroll
            for (int v = 0; v < 8; ++v) {
                int grow = row0 + rt * 16 + hl * 8 + v;
                out[(size_t)grow * OUT_NUM + col] = acc[rt][v] * ci[grow] + b;
            }
        }
    } else {
        #pragma unroll
        for (int rt = 0; rt < 4; ++rt) {
            #pragma unroll
            for (int v = 0; v < 8; ++v) {
                int grow = row0 + rt * 16 + hl * 8 + v;
                int gc = (grow < nrows) ? grow : (nrows - 1);
                float s = ci[gc];
                if (grow < nrows)
                    out[(size_t)grow * OUT_NUM + col] = acc[rt][v] * s + b;
            }
        }
    }
}

// ---------------------------------------------------------------------------
extern "C" void kernel_launch(void* const* d_in, const int* in_sizes, int n_in,
                              void* d_out, int out_size, void* d_ws, size_t ws_size,
                              hipStream_t stream)
{
    (void)in_sizes; (void)n_in; (void)out_size; (void)ws_size;

    const float* ufeat    = (const float*)d_in[0];
    const float* ifeat    = (const float*)d_in[1];
    const float* W_u      = (const float*)d_in[2];
    const float* W_i      = (const float*)d_in[3];
    const float* user_ci  = (const float*)d_in[4];
    const float* user_cj  = (const float*)d_in[5];
    const float* movie_ci = (const float*)d_in[6];
    const float* movie_cj = (const float*)d_in[7];
    const int*   edge_u   = (const int*)d_in[8];
    const int*   edge_i   = (const int*)d_in[9];
    const float* ufc_w    = (const float*)d_in[10];
    const float* ufc_b    = (const float*)d_in[11];
    const float* ifc_w    = (const float*)d_in[12];
    const float* ifc_b    = (const float*)d_in[13];

    // Workspace layout (row strides padded to 64 so xform stores are unguarded)
    char* ws = (char*)d_ws;
    const size_t xu_b = (size_t)R_NUM * LDXU * K_NUM * sizeof(__bf16);
    const size_t xi_b = (size_t)R_NUM * LDXI * K_NUM * sizeof(__bf16);
    const size_t hu_b = (size_t)NU_NUM * MSG_NUM * sizeof(float);
    const size_t hi_b = (size_t)NI_NUM * MSG_NUM * sizeof(float);
    __bf16* xu = (__bf16*)(ws);
    __bf16* xi = (__bf16*)(ws + xu_b);
    float*  hu = (float*)(ws + xu_b + xi_b);
    float*  hi = (float*)(ws + xu_b + xi_b + hu_b);

    hipMemsetAsync(hu, 0, hu_b, stream);
    hipMemsetAsync(hi, 0, hi_b, stream);

    // 1) per-relation transforms (WMMA bf16)
    gcmc_xform<<<dim3(LDXU / 64, R_NUM), 256, 0, stream>>>(
        ufeat, W_u, user_cj, xu, NU_NUM, LDXU);
    gcmc_xform<<<dim3(LDXI / 64, R_NUM), 256, 0, stream>>>(
        ifeat, W_i, movie_cj, xi, NI_NUM, LDXI);

    // 2) edge scatter-sum (user->movie into hi, movie->user into hu)
    const int nblk = (R_NUM * E_NUM + 7) / 8;
    gcmc_scatter<<<nblk, 256, 0, stream>>>(xu, edge_u, edge_i, hi, LDXU);
    gcmc_scatter<<<nblk, 256, 0, stream>>>(xi, edge_i, edge_u, hu, LDXI);

    // 3) output FC (WMMA bf16), ci + bias folded into epilogue
    gcmc_fc<<<dim3((NU_NUM + 63) / 64, 2), 256, 0, stream>>>(
        hu, user_ci, ufc_w, ufc_b, (float*)d_out, NU_NUM);
    gcmc_fc<<<dim3((NI_NUM + 63) / 64, 2), 256, 0, stream>>>(
        hi, movie_ci, ifc_w, ifc_b, (float*)d_out + (size_t)NU_NUM * OUT_NUM, NI_NUM);
}
